// Module_58969900974314
// MI455X (gfx1250) — matrix operand order repaired and checked by reading, not compiled, not run on hardware
//
#include <hip/hip_runtime.h>

// ---------------------------------------------------------------------------
// Additive (concat-MLP) attention, restructured:
//   hq[b,:]   = Q[b,:] @ W1[0:D,:] + b1                 (tiny)
//   hk[b,l,:] = K[b,l,:] @ W1[D:2D,:]                   (big GEMM -> bf16 WMMA)
//   s[b,l]    = w2 . ReLU(hk + hq) + b2                 (fused epilogue)
//   out[b,:]  = sum_l softmax_beta(s) * V[b,l,:]
// B=256, L=512, D=256, BETA=0.5
//
// W1k^T strips are staged into LDS with GLOBAL_LOAD_ASYNC_TO_LDS_B128
// (ASYNCcnt-tracked DMA, no VGPR round-trip), double-buffered so the DMA for
// strip nt+1 overlaps the 8 chained v_wmma_f32_16x16x32_bf16 of strip nt.
// ---------------------------------------------------------------------------

#define BATCH 256
#define LSEQ  512
#define DDIM  256

typedef __attribute__((ext_vector_type(16))) __bf16         v16bf;
typedef __attribute__((ext_vector_type(16))) unsigned short v16us;
typedef __attribute__((ext_vector_type(8)))  float          v8f;

__device__ __forceinline__ unsigned short f2b(float f) {
    // fp32 -> bf16, round-to-nearest-even
    unsigned int u = __float_as_uint(f);
    u += 0x7fffu + ((u >> 16) & 1u);
    return (unsigned short)(u >> 16);
}

// ---------------------------------------------------------------------------
// Kernel 1: hq[b,d] = sum_e Q[b,e] * W1[e,d] + b1[d]
// ---------------------------------------------------------------------------
__global__ void hq_kernel(const float* __restrict__ Q,
                          const float* __restrict__ W1,
                          const float* __restrict__ b1,
                          float* __restrict__ hq) {
    int b = blockIdx.x;
    int d = threadIdx.x;
    const float* q = Q + (size_t)b * DDIM;
    float s = b1[d];
#pragma unroll 4
    for (int e = 0; e < DDIM; ++e)
        s = fmaf(q[e], W1[(size_t)e * DDIM + d], s);
    hq[(size_t)b * DDIM + d] = s;
}

// ---------------------------------------------------------------------------
// Kernel 2: W1kT[d,e] = bf16( W1[D+e, d] )   (row-major [D][D] ushort/bf16)
// ---------------------------------------------------------------------------
__global__ void w1kt_kernel(const float* __restrict__ W1,
                            unsigned short* __restrict__ W1kT) {
    int i = blockIdx.x * 256 + threadIdx.x;   // 65536 total
    int d = i & (DDIM - 1);
    int e = i >> 8;
    float v = W1[(size_t)(DDIM + e) * DDIM + d];   // coalesced over d
    W1kT[(size_t)d * DDIM + e] = f2b(v);
}

// ---------------------------------------------------------------------------
// Kernel 3: scores via bf16 WMMA GEMM + fused ReLU/w2 epilogue.
// Grid: (L/64, B), block = 128 (4 wave32). Wave w owns rows l0..l0+15.
// ---------------------------------------------------------------------------
__global__ void scores_kernel(const float* __restrict__ K,
                              const unsigned short* __restrict__ W1kT,
                              const float* __restrict__ hq,
                              const float* __restrict__ w2,
                              const float* __restrict__ b2,
                              float* __restrict__ scores) {
    // two strip buffers: 16 cols x 256 e bf16, row stride 264 halves (+8 pad)
    __shared__ __align__(16) unsigned short strip[2][16 * 264];
    __shared__ float hqb_s[DDIM];
    __shared__ float w2_s[DDIM];

    const int tid  = threadIdx.x;
    const int wave = tid >> 5;
    const int lane = tid & 31;
    const int b    = blockIdx.y;
    const int l0   = blockIdx.x * 64 + wave * 16;

    // stage hq[b,:] and w2 (first in-loop barrier covers these)
    for (int i = tid; i < DDIM; i += 128) {
        hqb_s[i] = hq[(size_t)b * DDIM + i];
        w2_s[i]  = w2[i];
    }

    const uint4* g4 = (const uint4*)W1kT;   // 32 uint4 per 256-bf16 row
    // Low 32 bits of a generic LDS pointer == byte offset inside the wave's
    // LDS allocation (ISA 10.2), i.e. exactly the async-op's dsaddr operand.
    const unsigned int lds_base0 = (unsigned int)(uintptr_t)&strip[0][0];
    const unsigned int lds_base1 = (unsigned int)(uintptr_t)&strip[1][0];

    auto issue_strip = [&](int buf, int nt) {
        unsigned int base = buf ? lds_base1 : lds_base0;
#pragma unroll
        for (int k = 0; k < 4; ++k) {          // 512 b128 chunks / 128 threads
            int idx = tid + k * 128;
            int row = idx >> 5;                // 0..15 (column of this strip)
            int c   = idx & 31;                // 16B chunk within row
            const uint4* g = g4 + (size_t)(nt * 16 + row) * 32 + c;
            unsigned int l = base + (unsigned int)(row * 528 + c * 16);
            asm volatile("global_load_async_to_lds_b128 %0, %1, off"
                         :: "v"(l), "v"((const void*)g) : "memory");
        }
    };

    // ---- Load & convert A fragments (16 rows x 256 K-dim) into registers ----
    // 16-bit A 16x32 layout: lanes 0-15 -> M=lane, K = {0..7,16..23}+32*ec
    //                        lanes 16-31 -> M=lane-16, K = {8..15,24..31}+32*ec
    const int m = lane & 15;
    const int h = lane >> 4;
    const float* krow = K + ((size_t)b * LSEQ + l0 + m) * DDIM;

    issue_strip(0, 0);   // DMA for first strip overlaps A-fragment conversion

    v16bf afr[8];
#pragma unroll
    for (int ec = 0; ec < 8; ++ec) {
        const float* kp = krow + ec * 32 + h * 8;
        float4 f0 = *(const float4*)(kp);
        float4 f1 = *(const float4*)(kp + 4);
        float4 f2 = *(const float4*)(kp + 16);
        float4 f3 = *(const float4*)(kp + 20);
        v16us t;
        t[0]  = f2b(f0.x); t[1]  = f2b(f0.y); t[2]  = f2b(f0.z); t[3]  = f2b(f0.w);
        t[4]  = f2b(f1.x); t[5]  = f2b(f1.y); t[6]  = f2b(f1.z); t[7]  = f2b(f1.w);
        t[8]  = f2b(f2.x); t[9]  = f2b(f2.y); t[10] = f2b(f2.z); t[11] = f2b(f2.w);
        t[12] = f2b(f3.x); t[13] = f2b(f3.y); t[14] = f2b(f3.z); t[15] = f2b(f3.w);
        afr[ec] = __builtin_bit_cast(v16bf, t);
    }

    const int n  = lane & 15;          // output column within N-tile (B operand)
    const int kb = h << 4;             // 0 or 16: unused; A/B fragments offset K by 8*h instead

    float partial[8];
#pragma unroll
    for (int v = 0; v < 8; ++v) partial[v] = 0.0f;

    int cur = 0;
    for (int nt = 0; nt < 16; ++nt) {
        // wait for this wave's DMA of strip[cur], then make it WG-visible
        asm volatile("s_wait_asynccnt 0x0" ::: "memory");
        __syncthreads();
        // kick off next strip into the other buffer; overlaps WMMA below.
        // (Safe: strip[cur] is only re-issued at nt+2, after every wave passed
        //  the nt+1 barrier, i.e. after all reads of strip[cur] completed.)
        if (nt < 15) issue_strip(cur ^ 1, nt + 1);

        const unsigned short* sp = strip[cur];
        v8f acc = {};
#pragma unroll
        for (int ec = 0; ec < 8; ++ec) {
            // B 32x16 layout: lane n holds column n, K = {0..7,16..23}+8*h
            const uint4* bp = (const uint4*)(sp + n * 264 + ec * 32 + (h << 3));
            union { uint4 q[2]; v16us u; } uu;
            uu.q[0] = bp[0];
            uu.q[1] = bp[2];
            v16bf bfrag = __builtin_bit_cast(v16bf, uu.u);
            acc = __builtin_amdgcn_wmma_f32_16x16x32_bf16(
                false, afr[ec], false, bfrag, (short)0, acc, false, false);
        }

        // epilogue: lane holds column d = nt*16+n, rows M = h*8 + v
        float hqv = hqb_s[nt * 16 + n];
        float w2v = w2_s[nt * 16 + n];
#pragma unroll
        for (int v = 0; v < 8; ++v)
            partial[v] += w2v * fmaxf(acc[v] + hqv, 0.0f);

        cur ^= 1;
    }

    // reduce over the 16 lanes that share the same row set
#pragma unroll
    for (int v = 0; v < 8; ++v) {
#pragma unroll
        for (int off = 1; off < 16; off <<= 1)
            partial[v] += __shfl_xor(partial[v], off, 32);
    }
    if ((lane & 15) == 0) {
        float b2v = b2[0];
        int mbase = h * 8;
#pragma unroll
        for (int v = 0; v < 8; ++v)
            scores[(size_t)b * LSEQ + l0 + mbase + v] = partial[v] + b2v;
    }
}

// ---------------------------------------------------------------------------
// Kernel 4: beta-smoothed masked softmax + weighted sum of V. One block per b.
// weights = e * exp((1-beta)*m) / denom^beta, beta = 0.5
// ---------------------------------------------------------------------------
__global__ void ctx_kernel(const float* __restrict__ scores,
                           const unsigned char* __restrict__ mask,
                           const float* __restrict__ V,
                           float* __restrict__ out) {
    __shared__ float red[256];
    __shared__ float wsh[LSEQ];

    const int b   = blockIdx.x;
    const int tid = threadIdx.x;
    const float NEG = -1e30f;

    const unsigned char m0b = mask[(size_t)b * LSEQ + tid];
    const unsigned char m1b = mask[(size_t)b * LSEQ + 256 + tid];
    float s0 = m0b ? NEG : scores[(size_t)b * LSEQ + tid];
    float s1 = m1b ? NEG : scores[(size_t)b * LSEQ + 256 + tid];

    // max reduce
    red[tid] = fmaxf(s0, s1);
    __syncthreads();
    for (int off = 128; off > 0; off >>= 1) {
        if (tid < off) red[tid] = fmaxf(red[tid], red[tid + off]);
        __syncthreads();
    }
    float m = red[0];
    __syncthreads();

    float e0 = m0b ? 0.0f : __expf(s0 - m);
    float e1 = m1b ? 0.0f : __expf(s1 - m);

    // sum reduce
    red[tid] = e0 + e1;
    __syncthreads();
    for (int off = 128; off > 0; off >>= 1) {
        if (tid < off) red[tid] += red[tid + off];
        __syncthreads();
    }
    float denom = red[0];

    float scale = __expf(0.5f * m) / sqrtf(denom);   // (1-beta)=0.5, denom^0.5
    wsh[tid]       = e0 * scale;
    wsh[tid + 256] = e1 * scale;
    __syncthreads();

    // context: thread owns column d = tid; V row-contiguous -> coalesced
    const float* vcol = V + (size_t)b * LSEQ * DDIM + tid;
    float acc = 0.0f;
#pragma unroll 4
    for (int l = 0; l < LSEQ; ++l)
        acc = fmaf(wsh[l], vcol[(size_t)l * DDIM], acc);
    out[(size_t)b * DDIM + tid] = acc;
}

// ---------------------------------------------------------------------------
extern "C" void kernel_launch(void* const* d_in, const int* in_sizes, int n_in,
                              void* d_out, int out_size, void* d_ws, size_t ws_size,
                              hipStream_t stream) {
    const float*         Q    = (const float*)d_in[0];
    const float*         K    = (const float*)d_in[1];
    const float*         V    = (const float*)d_in[2];
    const unsigned char* mask = (const unsigned char*)d_in[3]; // jnp.bool_ = 1 byte
    const float*         W1   = (const float*)d_in[4];
    const float*         b1   = (const float*)d_in[5];
    const float*         w2   = (const float*)d_in[6];
    const float*         b2   = (const float*)d_in[7];
    float*               out  = (float*)d_out;

    char* ws = (char*)d_ws;
    float*          hq     = (float*)ws;                              // 256 KB
    float*          scores = (float*)(ws + 262144);                   // 512 KB
    unsigned short* w1kt   = (unsigned short*)(ws + 262144 + 524288); // 128 KB

    hq_kernel  <<<BATCH, DDIM, 0, stream>>>(Q, W1, b1, hq);
    w1kt_kernel<<<DDIM,  256,  0, stream>>>(W1, w1kt);
    scores_kernel<<<dim3(LSEQ / 64, BATCH), 128, 0, stream>>>(K, w1kt, hq, w2, b2, scores);
    ctx_kernel <<<BATCH, 256, 0, stream>>>(scores, mask, V, out);
}
